// CVRP_Decoder_13108240188002
// MI455X (gfx1250) — matrix operand order repaired and checked by reading, not compiled, not run on hardware
//
#include <hip/hip_runtime.h>

// ---------------------------------------------------------------------------
// Problem constants (match reference)
// ---------------------------------------------------------------------------
#define BB   32
#define NN   512
#define PP   512
#define EMBD 256
#define HH   16
#define DD   16
#define LOCALK 100

typedef __attribute__((ext_vector_type(8)))  __bf16 v8bf;
typedef __attribute__((ext_vector_type(16))) __bf16 v16bf;
typedef __attribute__((ext_vector_type(8)))  float  v8f;

static __device__ __forceinline__ __bf16 to_bf(float f) {
  unsigned u = __float_as_uint(f);
  unsigned r = (u + 0x7FFFu + ((u >> 16) & 1u)) >> 16;   // RNE
  return __builtin_bit_cast(__bf16, (unsigned short)r);
}

static __device__ __forceinline__ v8f wmma_bf16(v16bf a, v16bf b, v8f c) {
  return __builtin_amdgcn_wmma_f32_16x16x32_bf16(false, a, false, b,
                                                 (short)0, c, false, false);
}

// ---------------------------------------------------------------------------
// f32 -> bf16 conversion (optionally strided source rows)
// ---------------------------------------------------------------------------
__global__ void cvt2d(const float* __restrict__ src, __bf16* __restrict__ dst,
                      int rows, int cols, int srcStride) {
  int total = rows * cols;
  for (int idx = blockIdx.x * blockDim.x + threadIdx.x; idx < total;
       idx += gridDim.x * blockDim.x) {
    int r = idx / cols, c = idx - r * cols;
    dst[idx] = to_bf(src[(size_t)r * srcStride + c]);
  }
}

// ---------------------------------------------------------------------------
// Generic 16x64-per-wave GEMM:  Out[g, c] = sum_k A[g,k] * W[c,k]
//   A: [R,256] bf16 row-major, W: [256,256] bf16 ("[col][k]", K contiguous)
//   mode 0: plain row-major bf16 out             (K projection)
//   mode 1: out transposed per-head Vt[B,H,D,N]  (V projection, packed stores)
//   mode 2: + vecRow[g] * vecColW[c*257]         (Q projection, 'load' column)
//   mode 3: + bias[c]                            (Wc projection)
// ---------------------------------------------------------------------------
__global__ void gemm16x64(const __bf16* __restrict__ A,
                          const __bf16* __restrict__ W,
                          int mode,
                          __bf16* __restrict__ outRM,
                          __bf16* __restrict__ outVt,
                          const float* __restrict__ vecRow,
                          const float* __restrict__ vecColW,
                          const float* __restrict__ bias) {
  const int lane = threadIdx.x & 31;
  const int hv   = lane >> 4;
  const int l16  = lane & 15;
  const int rt   = blockIdx.x >> 2;
  const int c0   = (blockIdx.x & 3) * 64;
  const int g0   = rt * 16;

  // Cache all 8 K-chunk A-fragments (row = l16 of the tile)
  const __bf16* arow = A + (size_t)(g0 + l16) * EMBD;
  v16bf afr[8];
#pragma unroll
  for (int kc = 0; kc < 8; ++kc) {
    v8bf lo = *(const v8bf*)(arow + kc * 32 + hv * 16);
    v8bf hi = *(const v8bf*)(arow + kc * 32 + hv * 16 + 8);
#pragma unroll
    for (int i = 0; i < 8; ++i) { afr[kc][i] = lo[i]; afr[kc][8 + i] = hi[i]; }
  }

#pragma unroll
  for (int ct = 0; ct < 4; ++ct) {
    const int c = c0 + ct * 16 + l16;           // this lane's output column
    const __bf16* brow = W + (size_t)c * EMBD;  // K contiguous
    v8f acc = {};
#pragma unroll
    for (int kc = 0; kc < 8; ++kc) {
      v16bf bf = *(const v16bf*)(brow + kc * 32 + hv * 16);
      acc = wmma_bf16(afr[kc], bf, acc);
    }

    if (mode == 1) {
      // 8 accumulator elements are 8 consecutive n in Vt[B,H,D,N]: 1x16B store
      const int h = c >> 4, d = c & 15;
      const int bb = g0 >> 9, nb = (g0 & 511) + hv * 8;
      v8bf vv;
#pragma unroll
      for (int r = 0; r < 8; ++r) vv[r] = to_bf(acc[r]);
      *(v8bf*)(outVt + ((size_t)(bb * HH + h) * DD + d) * NN + nb) = vv;
    } else {
      float wqc = 0.0f, colAdd = 0.0f;
      if (mode == 2) wqc = vecColW[(size_t)c * 257];
      if (mode == 3) colAdd = bias[c];
#pragma unroll
      for (int r = 0; r < 8; ++r) {
        const int g = g0 + r + hv * 8;          // global row of this element
        float v = acc[r] + colAdd;
        if (mode == 2) v += vecRow[g] * wqc;
        outRM[(size_t)g * 256 + c] = to_bf(v);
      }
    }
  }
}

// ---------------------------------------------------------------------------
// Fused attention: per (b, h, 16-row P tile): S = QK^T/4 + mask, softmax,
// O = S V.  D=16 padded to K=32 in registers for the QK^T WMMA.
// ---------------------------------------------------------------------------
__global__ void attn_kernel(const __bf16* __restrict__ Qbf,
                            const __bf16* __restrict__ Kbf,
                            const __bf16* __restrict__ Vt,
                            const float* __restrict__ mask,
                            __bf16* __restrict__ attnOut) {
  __shared__ __align__(16) float sc[16 * NN];   // 32 KB score tile

  const int id  = blockIdx.x;
  const int pt  = id & 31;
  const int h   = (id >> 5) & 15;
  const int b   = id >> 9;
  const int p0  = pt * 16;
  const int lane = threadIdx.x & 31;
  const int hv = lane >> 4, l16 = lane & 15;

  // Early prefetch of the 32KB contiguous mask tile (consumed per score tile)
  {
    const char* mbase = (const char*)(mask + (size_t)(b * PP + p0) * NN);
#pragma unroll
    for (int i = 0; i < 16; ++i)
      __builtin_prefetch(mbase + i * 2048 + lane * 64, 0, 0);
  }

  // Q fragment, K=16 real + 16 zero pad
  const __bf16* qrow = Qbf + (size_t)(b * PP + p0 + l16) * 256 + h * 16;
  v8bf qlo = *(const v8bf*)(qrow);
  v8bf qhi = *(const v8bf*)(qrow + 8);
  v16bf afrag;
#pragma unroll
  for (int i = 0; i < 8; ++i) { afrag[i] = hv == 0 ? qlo[i] : to_bf(0.0f); afrag[8 + i] = hv == 0 ? qhi[i] : to_bf(0.0f); }

  // ---- scores ----
  for (int nt = 0; nt < 32; ++nt) {
    const int n0 = nt * 16;
    v16bf bfrag;
    if (hv == 0) {
      bfrag = *(const v16bf*)(Kbf + (size_t)(b * NN + n0 + l16) * 256 + h * 16);
    } else {
      v16bf z = {};
      bfrag = z;                                // K = 16..31 padding
    }
    v8f c = {};
    c = wmma_bf16(afrag, bfrag, c);
#pragma unroll
    for (int r = 0; r < 8; ++r) {
      const int m = r + hv * 8;
      const int n = n0 + l16;
      sc[m * NN + n] = c[r] * 0.25f +
                       mask[(size_t)(b * PP + p0 + m) * NN + n];
    }
  }
  __syncthreads();

  // ---- softmax (2 lanes per row, float4 scans) ----
  const int row  = l16;
  const int base = row * NN + hv * 256;
  {
    const float4* s4 = (const float4*)(sc + base);
    float mx = -3.0e38f;
    for (int i = 0; i < 64; ++i) {
      float4 v = s4[i];
      mx = fmaxf(mx, fmaxf(fmaxf(v.x, v.y), fmaxf(v.z, v.w)));
    }
    mx = fmaxf(mx, __shfl_xor(mx, 16));
    float4* s4w = (float4*)(sc + base);
    float sum = 0.0f;
    for (int i = 0; i < 64; ++i) {
      float4 v = s4w[i];
      v.x = __expf(v.x - mx); v.y = __expf(v.y - mx);
      v.z = __expf(v.z - mx); v.w = __expf(v.w - mx);
      s4w[i] = v;
      sum += (v.x + v.y) + (v.z + v.w);
    }
    sum += __shfl_xor(sum, 16);
    const float inv = 1.0f / sum;
    __syncthreads();

    // ---- O = softmax(S) * V ----
    v8f o = {};
    for (int nt = 0; nt < 16; ++nt) {
      const int n0 = nt * 32;
      const float4 a0 = *(const float4*)(sc + row * NN + n0 + hv * 16);
      const float4 a1 = *(const float4*)(sc + row * NN + n0 + hv * 16 + 4);
      const float4 a2 = *(const float4*)(sc + row * NN + n0 + hv * 16 + 8);
      const float4 a3 = *(const float4*)(sc + row * NN + n0 + hv * 16 + 12);
      v16bf af;
      af[0]  = to_bf(a0.x * inv); af[1]  = to_bf(a0.y * inv);
      af[2]  = to_bf(a0.z * inv); af[3]  = to_bf(a0.w * inv);
      af[4]  = to_bf(a1.x * inv); af[5]  = to_bf(a1.y * inv);
      af[6]  = to_bf(a1.z * inv); af[7]  = to_bf(a1.w * inv);
      af[8]  = to_bf(a2.x * inv); af[9]  = to_bf(a2.y * inv);
      af[10] = to_bf(a2.z * inv); af[11] = to_bf(a2.w * inv);
      af[12] = to_bf(a3.x * inv); af[13] = to_bf(a3.y * inv);
      af[14] = to_bf(a3.z * inv); af[15] = to_bf(a3.w * inv);
      v16bf bf = *(const v16bf*)(Vt + ((size_t)(b * HH + h) * DD + l16) * NN +
                                 n0 + hv * 16);
      o = wmma_bf16(af, bf, o);
    }
#pragma unroll
    for (int r = 0; r < 8; ++r) {
      const int m = r + hv * 8;
      attnOut[(size_t)(b * PP + p0 + m) * 256 + h * 16 + l16] = to_bf(o[r]);
    }
  }
}

// ---------------------------------------------------------------------------
// Fused final stage per (b, 16-row P tile):
//   score2 = MH @ E^T / 16 ; exact 100th-smallest-dist threshold (bitwise
//   binary search); penalty; 10*tanh; +mask; softmax -> probs
//   cur_dist tile staged with CDNA5 async global->LDS DMA (ASYNCcnt).
// ---------------------------------------------------------------------------
__global__ void final_kernel(const __bf16* __restrict__ MHbf,
                             const __bf16* __restrict__ Ebf,
                             const float* __restrict__ cur_dist,
                             const float* __restrict__ mask,
                             float* __restrict__ probs) {
  __shared__ __align__(16) float sc[16 * NN];   // 32 KB scores
  __shared__ __align__(16) float sd[16 * NN];   // 32 KB distances

  const int id = blockIdx.x;
  const int pt = id & 31;
  const int b  = id >> 5;
  const int p0 = pt * 16;
  const int lane = threadIdx.x & 31;
  const int hv = lane >> 4, l16 = lane & 15;

  // ---- stage cur_dist tile (16*512 contiguous f32) into LDS ----
#if defined(__gfx1250__)
  {
    // Async DMA: overlaps with A-frag loads and the whole score2 GEMM.
    const unsigned long long gbase =
        (unsigned long long)(uintptr_t)(cur_dist + (size_t)(b * PP + p0) * NN);
    const unsigned lbase = (unsigned)(uintptr_t)sd;  // generic->LDS: low bits
    for (int i = lane; i < (16 * NN) / 4; i += 32) {
      const unsigned la = lbase + (unsigned)i * 16u;
      const unsigned long long ga = gbase + (unsigned long long)i * 16u;
      asm volatile("global_load_async_to_lds_b128 %0, %1, off"
                   :: "v"(la), "v"(ga) : "memory");
    }
  }
#else
  {
    const float4* dsrc = (const float4*)(cur_dist + (size_t)(b * PP + p0) * NN);
    float4* ddst = (float4*)sd;
    for (int i = lane; i < (16 * NN) / 4; i += 32) ddst[i] = dsrc[i];
  }
#endif

  // Prefetch the mask tile (needed only after GEMM + selection)
  {
    const char* mbase = (const char*)(mask + (size_t)(b * PP + p0) * NN);
#pragma unroll
    for (int i = 0; i < 16; ++i)
      __builtin_prefetch(mbase + i * 2048 + lane * 64, 0, 0);
  }

  // Cache all 8 A-fragments of MH rows
  const __bf16* arow = MHbf + (size_t)(b * PP + p0 + l16) * EMBD;
  v16bf afr[8];
#pragma unroll
  for (int kc = 0; kc < 8; ++kc) {
    v8bf lo = *(const v8bf*)(arow + kc * 32 + hv * 16);
    v8bf hi = *(const v8bf*)(arow + kc * 32 + hv * 16 + 8);
#pragma unroll
    for (int i = 0; i < 8; ++i) { afr[kc][i] = lo[i]; afr[kc][8 + i] = hi[i]; }
  }

  // score2 tiles
  for (int nt = 0; nt < 32; ++nt) {
    const int n0 = nt * 16;
    const __bf16* brow = Ebf + (size_t)(b * NN + n0 + l16) * EMBD;
    v8f c = {};
#pragma unroll
    for (int kc = 0; kc < 8; ++kc) {
      v16bf bf = *(const v16bf*)(brow + kc * 32 + hv * 16);
      c = wmma_bf16(afr[kc], bf, c);
    }
#pragma unroll
    for (int r = 0; r < 8; ++r)
      sc[(r + hv * 8) * NN + n0 + l16] = c[r] * 0.0625f;   // /sqrt(EMB)
  }
#if defined(__gfx1250__)
  asm volatile("s_wait_asynccnt 0x0" ::: "memory");        // dist tile landed
#endif
  __syncthreads();

  // 100th-smallest distance per row: binary search on positive-float bits
  const int row  = l16;
  const int base = row * NN + hv * 256;
  const float4* d4 = (const float4*)(sd + base);
  unsigned lo = 0u, hi = 0x3F800000u;           // [0, 1.0f)
  for (int it = 0; it < 31; ++it) {
    const unsigned mid = (lo + hi) >> 1;
    const float mf = __uint_as_float(mid);
    int cnt = 0;
    for (int i = 0; i < 64; ++i) {
      const float4 v = d4[i];
      cnt += (v.x <= mf) + (v.y <= mf) + (v.z <= mf) + (v.w <= mf);
    }
    cnt += __shfl_xor(cnt, 16);
    if (cnt >= LOCALK) hi = mid; else lo = mid;
  }
  const float thr = __uint_as_float(hi);

  // penalty + clip + mask, track row max (float4 throughout)
  const size_t grow = (size_t)(b * PP + p0 + row) * NN + hv * 256;
  const float4* m4 = (const float4*)(mask + grow);
  float4* s4 = (float4*)(sc + base);
  float mx = -3.0e38f;
  for (int i = 0; i < 64; ++i) {
    float4 s = s4[i];
    const float4 d = d4[i];
    const float4 m = m4[i];
    s.x += (d.x <= thr) ? (-d.x * 0.70710678118f) : 1.0f;
    s.y += (d.y <= thr) ? (-d.y * 0.70710678118f) : 1.0f;
    s.z += (d.z <= thr) ? (-d.z * 0.70710678118f) : 1.0f;
    s.w += (d.w <= thr) ? (-d.w * 0.70710678118f) : 1.0f;
    s.x = 10.0f * tanhf(s.x) + m.x;
    s.y = 10.0f * tanhf(s.y) + m.y;
    s.z = 10.0f * tanhf(s.z) + m.z;
    s.w = 10.0f * tanhf(s.w) + m.w;
    s4[i] = s;
    mx = fmaxf(mx, fmaxf(fmaxf(s.x, s.y), fmaxf(s.z, s.w)));
  }
  mx = fmaxf(mx, __shfl_xor(mx, 16));
  float sum = 0.0f;
  for (int i = 0; i < 64; ++i) {
    float4 v = s4[i];
    v.x = __expf(v.x - mx); v.y = __expf(v.y - mx);
    v.z = __expf(v.z - mx); v.w = __expf(v.w - mx);
    s4[i] = v;
    sum += (v.x + v.y) + (v.z + v.w);
  }
  sum += __shfl_xor(sum, 16);
  const float inv = 1.0f / sum;
  float4* p4 = (float4*)(probs + grow);
  for (int i = 0; i < 64; ++i) {
    float4 v = s4[i];
    v.x *= inv; v.y *= inv; v.z *= inv; v.w *= inv;
    p4[i] = v;
  }
}

// ---------------------------------------------------------------------------
// Host launcher
// ---------------------------------------------------------------------------
extern "C" void kernel_launch(void* const* d_in, const int* in_sizes, int n_in,
                              void* d_out, int out_size, void* d_ws, size_t ws_size,
                              hipStream_t stream) {
  (void)in_sizes; (void)n_in; (void)out_size; (void)ws_size;
  const float* encoded_nodes = (const float*)d_in[0];
  const float* encoded_last  = (const float*)d_in[1];
  const float* loadv         = (const float*)d_in[2];
  const float* cur_dist      = (const float*)d_in[3];
  // d_in[4] cur_theta, d_in[5] ins_feature: unused in 'dis' bias path
  const float* ninf_mask     = (const float*)d_in[6];
  const float* Wq_last_w     = (const float*)d_in[7];
  const float* Wk_w          = (const float*)d_in[8];
  const float* Wv_w          = (const float*)d_in[9];
  const float* Wc_w          = (const float*)d_in[10];
  const float* Wc_b          = (const float*)d_in[11];
  float* probs = (float*)d_out;

  char* ws = (char*)d_ws;
  size_t off = 0;
  auto alloc = [&](size_t bytes) -> char* {
    char* p = ws + off;
    off += (bytes + 255) & ~(size_t)255;
    return p;
  };
  const size_t bigBf = (size_t)BB * PP * EMBD * sizeof(__bf16);   // 8 MB each
  __bf16* Ebf    = (__bf16*)alloc(bigBf);   // encoded_nodes bf16 [B*N,256]
  __bf16* Lbf    = (__bf16*)alloc(bigBf);   // encoded_last  bf16 [B*P,256]
  __bf16* Kbf    = (__bf16*)alloc(bigBf);   // K proj        [B*N,256]
  __bf16* Vt     = (__bf16*)alloc(bigBf);   // V proj transposed [B,H,D,N]
  __bf16* Qbf    = (__bf16*)alloc(bigBf);   // Q proj        [B*P,256]
  __bf16* ATTNbf = (__bf16*)alloc(bigBf);   // attention out [B*P,256]
  __bf16* MHbf   = (__bf16*)alloc(bigBf);   // Wc proj out   [B*P,256]
  __bf16* WkBf   = (__bf16*)alloc(256 * 256 * sizeof(__bf16));
  __bf16* WvBf   = (__bf16*)alloc(256 * 256 * sizeof(__bf16));
  __bf16* WqBf   = (__bf16*)alloc(256 * 256 * sizeof(__bf16));
  __bf16* WcBf   = (__bf16*)alloc(256 * 256 * sizeof(__bf16));

  // bf16 conversions
  cvt2d<<<4096, 256, 0, stream>>>(encoded_nodes, Ebf, BB * NN, EMBD, EMBD);
  cvt2d<<<4096, 256, 0, stream>>>(encoded_last,  Lbf, BB * PP, EMBD, EMBD);
  cvt2d<<<64, 256, 0, stream>>>(Wk_w,      WkBf, 256, 256, 256);
  cvt2d<<<64, 256, 0, stream>>>(Wv_w,      WvBf, 256, 256, 256);
  cvt2d<<<64, 256, 0, stream>>>(Wq_last_w, WqBf, 256, 256, 257); // drop col 256
  cvt2d<<<64, 256, 0, stream>>>(Wc_w,      WcBf, 256, 256, 256);

  const int R = BB * NN;               // 16384 rows (P == N)
  // K = E @ Wk^T
  gemm16x64<<<(R / 16) * 4, 32, 0, stream>>>(Ebf, WkBf, 0, Kbf, nullptr,
                                             nullptr, nullptr, nullptr);
  // V = E @ Wv^T  (stored transposed per head)
  gemm16x64<<<(R / 16) * 4, 32, 0, stream>>>(Ebf, WvBf, 1, nullptr, Vt,
                                             nullptr, nullptr, nullptr);
  // Q = [last, load] @ Wq^T  (rank-1 update handles the 'load' column)
  gemm16x64<<<(R / 16) * 4, 32, 0, stream>>>(Lbf, WqBf, 2, Qbf, nullptr,
                                             loadv, Wq_last_w + 256, nullptr);
  // fused MHA
  attn_kernel<<<BB * HH * (PP / 16), 32, 0, stream>>>(Qbf, Kbf, Vt, ninf_mask,
                                                      ATTNbf);
  // MH = attn @ Wc^T + b
  gemm16x64<<<(R / 16) * 4, 32, 0, stream>>>(ATTNbf, WcBf, 3, MHbf, nullptr,
                                             nullptr, nullptr, Wc_b);
  // fused scoring / penalty / softmax
  final_kernel<<<BB * (PP / 16), 32, 0, stream>>>(MHbf, Ebf, cur_dist,
                                                  ninf_mask, probs);
}
